// SelfAttention_17824114278575
// MI455X (gfx1250) — matrix operand order repaired and checked by reading, not compiled, not run on hardware
//
#include <hip/hip_runtime.h>
#include <hip/hip_bf16.h>

#define HIDDEN 768
#define SEQ    4096
#define BATCH  4
#define KT     32            // keys per flash-attention step (= WMMA K for bf16)

typedef __attribute__((ext_vector_type(16))) __bf16 v16bf;
typedef __attribute__((ext_vector_type(8)))  __bf16 v8bf;
typedef __attribute__((ext_vector_type(8)))  float  v8f;
typedef __attribute__((ext_vector_type(4)))  unsigned int v4u;
typedef __attribute__((ext_vector_type(4)))  int    v4i;
typedef __attribute__((ext_vector_type(8)))  int    v8i;

#if defined(__has_builtin)
#if __has_builtin(__builtin_amdgcn_tensor_load_to_lds) && \
    __has_builtin(__builtin_amdgcn_s_wait_tensorcnt)
#define HAVE_TDM 1
#endif
#endif
#ifndef HAVE_TDM
#define HAVE_TDM 0
#endif

// ---------------------------------------------------------------------------
// WMMA helpers (CDNA5 16x16x32 bf16, fp32 accumulate)
// ---------------------------------------------------------------------------
__device__ __forceinline__ v8f wmma_bf16(v16bf a, v16bf b, v8f c) {
  // (neg_a, A, neg_b, B, c_mod, C, reuse_a, reuse_b)
  return __builtin_amdgcn_wmma_f32_16x16x32_bf16(false, a, false, b, (short)0, c,
                                                 false, false);
}

// A fragment (16x32, 16-bit): lane m=lane&15 holds row m; per lane-half h:
// elements 0..7 <-> K = k0 + 8h + j, elements 8..15 <-> K = k0 + 16 + 8h + j.
__device__ __forceinline__ v16bf load_frag_a(const __bf16* base, int ld, int k0) {
  int lane = threadIdx.x & 31;
  int m = lane & 15, h = lane >> 4;
  const __bf16* p = base + (size_t)m * ld + k0 + 8 * h;
  v8bf lo = *(const v8bf*)p;
  v8bf hi = *(const v8bf*)(p + 16);
  v16bf a;
#pragma unroll
  for (int i = 0; i < 8; ++i) { a[i] = lo[i]; a[i + 8] = hi[i]; }
  return a;
}

// B fragment (32x16, 16-bit): lane n=lane&15 holds column n; per lane-half h:
// elements 0..7 <-> K = k0 + 8h + j, elements 8..15 <-> K = k0 + 16 + 8h + j.  Columns must be contiguous in memory
// (column n starts at base + n*ld).
__device__ __forceinline__ v16bf load_frag_b(const __bf16* base, int ld, int k0) {
  int lane = threadIdx.x & 31;
  int n = lane & 15, h = lane >> 4;
  const __bf16* p = base + (size_t)n * ld + k0 + 8 * h;
  v8bf lo = *(const v8bf*)p;
  v8bf hi = *(const v8bf*)(p + 16);
  v16bf b;
#pragma unroll
  for (int i = 0; i < 8; ++i) { b[i] = lo[i]; b[i + 8] = hi[i]; }
  return b;
}

// ---------------------------------------------------------------------------
// Tensor Data Mover: build D# groups per CDNA5 ISA and
// issue tensor_load_to_lds.  2-D tile (rows x cols), elements = 2 bytes.
//   lds_off       : byte offset of destination within workgroup LDS
//   gaddr         : global address of tile start
//   tensor_d0/d1  : tensor extents (elements / rows) for OOB
//   stride0       : elements between consecutive dim-1 rows
//   tile_d0/d1    : tile extents (tile_d1 == 0 -> 1-D transfer)
// ---------------------------------------------------------------------------
#if HAVE_TDM
__device__ __forceinline__ void tdm_load_2d_bf16(
    unsigned int lds_off, const void* gaddr, unsigned int tensor_d0,
    unsigned int tensor_d1, unsigned long long stride0, unsigned int tile_d0,
    unsigned int tile_d1) {
  unsigned long long ga = (unsigned long long)gaddr;
  v4u g0;
  g0[0] = 1u;                                        // count=1, user D#
  g0[1] = lds_off;                                   // lds_addr
  g0[2] = (unsigned int)(ga & 0xffffffffu);          // global_addr[31:0]
  g0[3] = (unsigned int)((ga >> 32) & 0x01ffffffu) | (2u << 30); // type=2
  v8i g1;
  g1[0] = (int)(1u << 16);                           // data_size=1 (2 bytes)
  g1[1] = (int)((tensor_d0 & 0xffffu) << 16);        // tensor_dim0[15:0]
  g1[2] = (int)((tensor_d0 >> 16) | ((tensor_d1 & 0xffffu) << 16));
  g1[3] = (int)((tensor_d1 >> 16) | ((tile_d0 & 0xffffu) << 16));
  g1[4] = (int)(tile_d1 & 0xffffu);                  // tile_dim1 (0 = 1-D)
  g1[5] = (int)(stride0 & 0xffffffffu);              // tensor_dim0_stride
  g1[6] = (int)((stride0 >> 32) & 0xffffu);
  g1[7] = 0;
  v4i gz = {0, 0, 0, 0};
#if __clang_major__ >= 23
  v8i gz8 = {0, 0, 0, 0, 0, 0, 0, 0};
  __builtin_amdgcn_tensor_load_to_lds(g0, g1, gz, gz, gz8, 0);
#else
  __builtin_amdgcn_tensor_load_to_lds(g0, g1, gz, gz, 0);
#endif
}
#endif

// ---------------------------------------------------------------------------
// Kernel 1: LayerNorm (fp32 in) -> bf16 normalized activations
// ---------------------------------------------------------------------------
__global__ __launch_bounds__(256) void ln_bf16_kernel(
    const float* __restrict__ x, const float* __restrict__ w,
    const float* __restrict__ b, __bf16* __restrict__ out) {
  int row = blockIdx.x;
  const float* xr = x + (size_t)row * HIDDEN;
  int t = threadIdx.x;
  float v0 = xr[t], v1 = xr[t + 256], v2 = xr[t + 512];
  float s = v0 + v1 + v2;
  float sq = v0 * v0 + v1 * v1 + v2 * v2;
#pragma unroll
  for (int m = 1; m < 32; m <<= 1) {
    s  += __shfl_xor(s, m);
    sq += __shfl_xor(sq, m);
  }
  __shared__ float ssum[8], ssq[8];
  int wid = t >> 5;
  if ((t & 31) == 0) { ssum[wid] = s; ssq[wid] = sq; }
  __syncthreads();
  float ts = 0.f, tq = 0.f;
#pragma unroll
  for (int i = 0; i < 8; ++i) { ts += ssum[i]; tq += ssq[i]; }
  float mean = ts * (1.0f / HIDDEN);
  float var  = tq * (1.0f / HIDDEN) - mean * mean;
  float rstd = rsqrtf(var + 1e-12f);
  __bf16* orow = out + (size_t)row * HIDDEN;
  orow[t]       = (__bf16)(w[t] * ((v0 - mean) * rstd) + b[t]);
  orow[t + 256] = (__bf16)(w[t + 256] * ((v1 - mean) * rstd) + b[t + 256]);
  orow[t + 512] = (__bf16)(w[t + 512] * ((v2 - mean) * rstd) + b[t + 512]);
}

// ---------------------------------------------------------------------------
// Kernel 2: weight transpose + cast: wT[z][out][in] = w_z[in][out] in bf16
// ---------------------------------------------------------------------------
__global__ __launch_bounds__(256) void wt_bf16_kernel(
    const float* __restrict__ wq, const float* __restrict__ wk,
    const float* __restrict__ wv, __bf16* __restrict__ wT) {
  int idx = blockIdx.x * 256 + threadIdx.x;       // 0 .. 768*768-1
  int m = blockIdx.y;                             // 0=q 1=k 2=v
  const float* src = (m == 0) ? wq : ((m == 1) ? wk : wv);
  int o = idx / HIDDEN, i = idx - o * HIDDEN;
  wT[(size_t)m * HIDDEN * HIDDEN + (size_t)o * HIDDEN + i] =
      (__bf16)src[(size_t)i * HIDDEN + o];
}

// ---------------------------------------------------------------------------
// Kernel 3: QKV projection GEMM via WMMA.
//   q,k row-major [B*S][768] bf16 ; v written transposed vT[b][d][s].
// ---------------------------------------------------------------------------
__global__ __launch_bounds__(128) void qkv_gemm_kernel(
    const __bf16* __restrict__ xb, const __bf16* __restrict__ wT,
    const float* __restrict__ bq, const float* __restrict__ bk,
    const float* __restrict__ bv, __bf16* __restrict__ qb,
    __bf16* __restrict__ kb, __bf16* __restrict__ vTb) {
  int wave = threadIdx.x >> 5;
  int lane = threadIdx.x & 31;
  int r0 = blockIdx.x * 16;                       // row tile (over B*S rows)
  int c0 = (blockIdx.y * 4 + wave) * 16;          // output-feature tile
  int z  = blockIdx.z;
  const __bf16* w = wT + (size_t)z * HIDDEN * HIDDEN + (size_t)c0 * HIDDEN;
  const __bf16* a0 = xb + (size_t)r0 * HIDDEN;

  v8f acc = {0.f, 0.f, 0.f, 0.f, 0.f, 0.f, 0.f, 0.f};
#pragma unroll
  for (int kk = 0; kk < HIDDEN; kk += 32) {
    v16bf a = load_frag_a(a0, HIDDEN, kk);
    v16bf b = load_frag_b(w, HIDDEN, kk);
    acc = wmma_bf16(a, b, acc);
  }

  const float* bias = (z == 0) ? bq : ((z == 1) ? bk : bv);
  int n = lane & 15, h = lane >> 4;
  int col = c0 + n;
  float bb = bias[col];
  if (z < 2) {
    __bf16* dst = (z == 0) ? qb : kb;
#pragma unroll
    for (int r = 0; r < 8; ++r) {
      int row = r0 + r + 8 * h;
      dst[(size_t)row * HIDDEN + col] = (__bf16)(acc[r] + bb);
    }
  } else {
#pragma unroll
    for (int r = 0; r < 8; ++r) {
      int row = r0 + r + 8 * h;
      int bidx = row >> 12;                       // row / SEQ
      int s = row & (SEQ - 1);
      vTb[((size_t)bidx * HIDDEN + col) * SEQ + s] = (__bf16)(acc[r] + bb);
    }
  }
}

// ---------------------------------------------------------------------------
// Kernel 4: flash attention with masked-threshold softmax + residual.
// block = 128 threads = 4 waves; each wave: 16 queries x full 768 features.
// K tile (32x768) and vT tile (768x32) staged into LDS by the Tensor Data
// Mover (TDM), issued by wave 0 and synchronized via TENSORcnt + barrier.
// ---------------------------------------------------------------------------
__global__ __launch_bounds__(128, 1) void attn_kernel(
    const __bf16* __restrict__ qb, const __bf16* __restrict__ kb,
    const __bf16* __restrict__ vTb, const float* __restrict__ mask,
    const float* __restrict__ input, float* __restrict__ out) {
  extern __shared__ char smem[];
  __bf16* sK  = (__bf16*)smem;                    // [KT][HIDDEN]   48 KB @ 0
  __bf16* sVT = sK + KT * HIDDEN;                 // [HIDDEN][KT]   48 KB @ 49152
  __bf16* sP  = sVT + HIDDEN * KT;                // [4 waves][16][KT]

  int wave = threadIdx.x >> 5;
  int lane = threadIdx.x & 31;
  int n = lane & 15, h = lane >> 4;
  int b = blockIdx.y;
  int q0 = (blockIdx.x * 4 + wave) * 16;          // query tile within batch
  const __bf16* Qbase = qb + ((size_t)b * SEQ + q0) * HIDDEN;
  __bf16* Pw = sP + wave * 16 * KT;

  v8f zero = {0.f, 0.f, 0.f, 0.f, 0.f, 0.f, 0.f, 0.f};
  v8f acc[48];
#pragma unroll
  for (int t = 0; t < 48; ++t) acc[t] = zero;
  float mrun[8], lrun[8];
#pragma unroll
  for (int r = 0; r < 8; ++r) { mrun[r] = -1e30f; lrun[r] = 0.f; }

  const float inv_scale = 0.125f;                 // 1/sqrt(HEAD_DIM=64)

  for (int k0 = 0; k0 < SEQ; k0 += KT) {
    __syncthreads();                              // prev-tile readers done
#if HAVE_TDM
    if (wave == 0) {
      // K tile: contiguous 32*768 bf16 (48 KB), 1-D transfer -> LDS offset 0
      tdm_load_2d_bf16(0u, kb + ((size_t)b * SEQ + k0) * HIDDEN,
                       (unsigned)(SEQ * HIDDEN), 1u,
                       (unsigned long long)(SEQ * HIDDEN),
                       (unsigned)(KT * HIDDEN), 0u);
      // vT tile: 768 rows x 32 keys, row stride SEQ -> LDS offset 49152
      tdm_load_2d_bf16((unsigned)(KT * HIDDEN * 2),
                       vTb + (size_t)b * HIDDEN * SEQ + k0, (unsigned)SEQ,
                       (unsigned)HIDDEN, (unsigned long long)SEQ, (unsigned)KT,
                       (unsigned)HIDDEN);
      __builtin_amdgcn_s_wait_tensorcnt(0);
    }
#else
    {
      const uint4* src = (const uint4*)(kb + ((size_t)b * SEQ + k0) * HIDDEN);
      uint4* dst = (uint4*)sK;
      for (int i = threadIdx.x; i < KT * HIDDEN / 8; i += 128) dst[i] = src[i];
    }
    for (int i = threadIdx.x; i < HIDDEN * (KT / 8); i += 128) {
      int d = i >> 2, part = i & 3;
      const uint4* s =
          (const uint4*)(vTb + ((size_t)b * HIDDEN + d) * SEQ + k0 + part * 8);
      *(uint4*)(sVT + d * KT + part * 8) = *s;
    }
#endif
    __syncthreads();                              // staging visible

    // --- scores: 16 queries x 32 keys, accumulate over 768 features ---
    v8f s0 = zero, s1 = zero;
#pragma unroll
    for (int kk = 0; kk < HIDDEN; kk += 32) {
      v16bf aq = load_frag_a(Qbase, HIDDEN, kk);             // L2-resident
      v16bf bk0 = load_frag_b(sK, HIDDEN, kk);               // keys k0..+15
      v16bf bk1 = load_frag_b(sK + 16 * HIDDEN, HIDDEN, kk); // keys +16..31
      s0 = wmma_bf16(aq, bk0, s0);
      s1 = wmma_bf16(aq, bk1, s1);
    }

    // --- mask * 1/8, threshold, online softmax ---
    float sc[8];
#pragma unroll
    for (int r = 0; r < 8; ++r) {
      size_t mi = ((size_t)b * SEQ + q0 + r + 8 * h) * (size_t)SEQ + k0 + n;
      float m0 = __builtin_nontemporal_load(&mask[mi]);      // streamed once
      float m1 = __builtin_nontemporal_load(&mask[mi + 16]);
      float x0 = s0[r] * m0 * inv_scale;
      float x1 = s1[r] * m1 * inv_scale;
      x0 = (x0 < 1e-3f) ? -1e4f : x0;
      x1 = (x1 < 1e-3f) ? -1e4f : x1;
      float tmax = fmaxf(x0, x1);
#pragma unroll
      for (int m = 1; m < 16; m <<= 1) tmax = fmaxf(tmax, __shfl_xor(tmax, m));
      float mnew = fmaxf(mrun[r], tmax);
      sc[r] = __expf(mrun[r] - mnew);
      mrun[r] = mnew;
      float p0 = __expf(x0 - mnew);
      float p1 = __expf(x1 - mnew);
      float ps = p0 + p1;
#pragma unroll
      for (int m = 1; m < 16; m <<= 1) ps += __shfl_xor(ps, m);
      lrun[r] = lrun[r] * sc[r] + ps;
      // probs to wave-private LDS, row-major [16][KT], for the A-fragment
      Pw[(r + 8 * h) * KT + n]      = (__bf16)p0;
      Pw[(r + 8 * h) * KT + n + 16] = (__bf16)p1;
    }
    // rescale accumulators
#pragma unroll
    for (int t = 0; t < 48; ++t)
#pragma unroll
      for (int r = 0; r < 8; ++r) acc[t][r] *= sc[r];

    // --- probs @ V : K = 32 keys, 48 feature tiles of 16 ---
    v16bf ap = load_frag_a(Pw, KT, 0);
#pragma unroll
    for (int t = 0; t < 48; ++t) {
      v16bf bv = load_frag_b(sVT + t * 16 * KT, KT, 0);
      acc[t] = wmma_bf16(ap, bv, acc[t]);
    }
  }

  // --- normalize, residual, store fp32 (all streaming -> non-temporal) ---
  float invl[8];
#pragma unroll
  for (int r = 0; r < 8; ++r) invl[r] = 1.0f / lrun[r];
#pragma unroll
  for (int t = 0; t < 48; ++t) {
    int d = t * 16 + n;
#pragma unroll
    for (int r = 0; r < 8; ++r) {
      size_t oi = ((size_t)b * SEQ + q0 + r + 8 * h) * HIDDEN + d;
      float res = __builtin_nontemporal_load(&input[oi]);
      __builtin_nontemporal_store(acc[t][r] * invl[r] + res, &out[oi]);
    }
  }
}

// ---------------------------------------------------------------------------
extern "C" void kernel_launch(void* const* d_in, const int* in_sizes, int n_in,
                              void* d_out, int out_size, void* d_ws,
                              size_t ws_size, hipStream_t stream) {
  (void)in_sizes; (void)n_in; (void)out_size; (void)ws_size;
  const float* x    = (const float*)d_in[0];
  const float* mask = (const float*)d_in[1];
  const float* lnw  = (const float*)d_in[2];
  const float* lnb  = (const float*)d_in[3];
  const float* wq   = (const float*)d_in[4];
  const float* bq   = (const float*)d_in[5];
  const float* wk   = (const float*)d_in[6];
  const float* bk   = (const float*)d_in[7];
  const float* wv   = (const float*)d_in[8];
  const float* bv   = (const float*)d_in[9];
  float* out = (float*)d_out;

  const size_t nrow = (size_t)BATCH * SEQ;
  __bf16* xb  = (__bf16*)d_ws;
  __bf16* qb  = xb + nrow * HIDDEN;
  __bf16* kb  = qb + nrow * HIDDEN;
  __bf16* vTb = kb + nrow * HIDDEN;
  __bf16* wT  = vTb + nrow * HIDDEN;               // 3*768*768 bf16

  wt_bf16_kernel<<<dim3((HIDDEN * HIDDEN) / 256, 3), 256, 0, stream>>>(wq, wk, wv, wT);
  ln_bf16_kernel<<<dim3((unsigned)nrow), 256, 0, stream>>>(x, lnw, lnb, xb);
  qkv_gemm_kernel<<<dim3((unsigned)(nrow / 16), HIDDEN / 64, 3), 128, 0, stream>>>(
      xb, wT, bq, bk, bv, qb, kb, vTb);

  size_t smem = (size_t)KT * HIDDEN * 2 + (size_t)HIDDEN * KT * 2 + 4 * 16 * KT * 2;
  attn_kernel<<<dim3(SEQ / 64, BATCH), 128, smem, stream>>>(qb, kb, vTb, mask, x, out);
}
